// RePoMultiHeadAttention_56264071578135
// MI455X (gfx1250) — matrix operand order repaired and checked by reading, not compiled, not run on hardware
//
#include <hip/hip_runtime.h>
#include <hip/hip_bf16.h>

typedef __bf16 bf16_t;
typedef __attribute__((ext_vector_type(16))) __bf16 v16bf;
typedef __attribute__((ext_vector_type(8)))  __bf16 v8bf;
typedef __attribute__((ext_vector_type(8)))  float  v8f;

#define HIDDEN   1024
#define HEADS    16
#define HEAD_DIM 64
#define POS_DIM  128
#define BATCH    2
#define SEQ      2048
#define TOKENS   (BATCH * SEQ)   // 4096

__device__ __forceinline__ v16bf concat8(v8bf a, v8bf b) {
    return __builtin_shufflevector(a, b, 0, 1, 2, 3, 4, 5, 6, 7,
                                         8, 9, 10, 11, 12, 13, 14, 15);
}

// ---------------------------------------------------------------------------
// f32 -> bf16 conversion, 8 elements/thread (2x b128 in, b128 out)
// ---------------------------------------------------------------------------
__global__ void f32_to_bf16_vec_kernel(const float* __restrict__ src,
                                       bf16_t* __restrict__ dst, int n8) {
    int i = blockIdx.x * blockDim.x + threadIdx.x;
    if (i >= n8) return;
    v8f v = *(const v8f*)(src + (size_t)i * 8);
    v8bf o;
#pragma unroll
    for (int j = 0; j < 8; ++j) o[j] = (bf16_t)v[j];
    *(v8bf*)(dst + (size_t)i * 8) = o;
}

// ---------------------------------------------------------------------------
// Tiled transposes (32x32 LDS tile, 256 threads). R, C multiples of 32.
// out(C x R) = in(R x C)^T
// ---------------------------------------------------------------------------
__global__ void transpose_f32_to_bf16_kernel(const float* __restrict__ in,
                                             bf16_t* __restrict__ out, int R, int C) {
    __shared__ float tile[32][33];
    int c0 = blockIdx.x * 32, r0 = blockIdx.y * 32;
    int tx = threadIdx.x & 31, ty = threadIdx.x >> 5;  // ty = 0..7
#pragma unroll
    for (int i = 0; i < 32; i += 8)
        tile[ty + i][tx] = in[(size_t)(r0 + ty + i) * C + c0 + tx];
    __syncthreads();
#pragma unroll
    for (int i = 0; i < 32; i += 8)
        out[(size_t)(c0 + ty + i) * R + r0 + tx] = (bf16_t)tile[tx][ty + i];
}

__global__ void transpose_bf16_kernel(const bf16_t* __restrict__ in,
                                      bf16_t* __restrict__ out, int R, int C) {
    __shared__ bf16_t tile[32][33];
    int c0 = blockIdx.x * 32, r0 = blockIdx.y * 32;
    int tx = threadIdx.x & 31, ty = threadIdx.x >> 5;
#pragma unroll
    for (int i = 0; i < 32; i += 8)
        tile[ty + i][tx] = in[(size_t)(r0 + ty + i) * C + c0 + tx];
    __syncthreads();
#pragma unroll
    for (int i = 0; i < 32; i += 8)
        out[(size_t)(c0 + ty + i) * R + r0 + tx] = tile[tx][ty + i];
}

// ---------------------------------------------------------------------------
// WMMA fragment helpers (wave32; contiguous k order, same for A and B), all b128 loads.
// A 16x32: lanes 0-15 row=lane, K 0..15; lanes 16-31 same rows,
//          K 16..31  -> 16 contiguous bf16 per lane.
// B 32x16: col = lane&15; lanes 0-15 K=0..15, lanes 16-31 K=16..31
//          -> from a K-contiguous (transposed) matrix: 16 contiguous bf16.
// C/D 16x16 f32: col = lane&15; VGPR r -> row r (lanes 0-15) / r+8 (16-31).
// ---------------------------------------------------------------------------
__device__ __forceinline__ v16bf load_a_frag(const bf16_t* __restrict__ A,
                                             int lda, int row0, int k0, int lane) {
    const bf16_t* p = A + (size_t)(row0 + (lane & 15)) * lda + k0 + ((lane >> 4) << 4);
    return concat8(*(const v8bf*)p, *(const v8bf*)(p + 8));
}

// Bt is N x K row-major (K contiguous)
__device__ __forceinline__ v16bf load_bt_frag(const bf16_t* __restrict__ Bt,
                                              int ldk, int k0, int col0, int lane) {
    const bf16_t* p = Bt + (size_t)(col0 + (lane & 15)) * ldk + k0 + ((lane >> 4) << 4);
    return concat8(*(const v8bf*)p, *(const v8bf*)(p + 8));
}

// ---------------------------------------------------------------------------
// GEMM: C(MxN) = A(MxK bf16) * B(KxN), B given transposed as Bt(NxK).
// block = 128 (4 waves stacked along M); each wave computes a 32x64 tile
// (2 A-frags x 4 B-frags = 8 WMMAs per k-step, 12 b128 loads -> 1.5 ld/WMMA).
// grid = (M/128, N/64).
// ---------------------------------------------------------------------------
__global__ void __launch_bounds__(128)
gemm_bf16_kernel(const bf16_t* __restrict__ A,
                 const bf16_t* __restrict__ Bt,
                 float* __restrict__ Cf, bf16_t* __restrict__ Cb,
                 int M, int N, int K) {
    int lane = threadIdx.x & 31;
    int wave = threadIdx.x >> 5;
    int row0 = blockIdx.x * 128 + wave * 32;
    int col0 = blockIdx.y * 64;
    v8f acc[2][4] = {};
    for (int k0 = 0; k0 < K; k0 += 32) {
        v16bf a0 = load_a_frag(A, K, row0,      k0, lane);
        v16bf a1 = load_a_frag(A, K, row0 + 16, k0, lane);
        v16bf b[4];
#pragma unroll
        for (int n = 0; n < 4; ++n)
            b[n] = load_bt_frag(Bt, K, k0, col0 + n * 16, lane);
#pragma unroll
        for (int n = 0; n < 4; ++n) {
            acc[0][n] = __builtin_amdgcn_wmma_f32_16x16x32_bf16(false, a0, false, b[n],
                                                                (short)0, acc[0][n], false, false);
            acc[1][n] = __builtin_amdgcn_wmma_f32_16x16x32_bf16(false, a1, false, b[n],
                                                                (short)0, acc[1][n], false, false);
        }
    }
    int c = col0 + (lane & 15);
    int rb = ((lane >> 4) << 3);
#pragma unroll
    for (int t = 0; t < 2; ++t) {
        int rbase = row0 + t * 16 + rb;
        if (Cf) {
#pragma unroll
            for (int r = 0; r < 8; ++r)
#pragma unroll
                for (int n = 0; n < 4; ++n)
                    Cf[(size_t)(rbase + r) * N + c + n * 16] = acc[t][n][r];
        } else {
#pragma unroll
            for (int r = 0; r < 8; ++r)
#pragma unroll
                for (int n = 0; n < 4; ++n)
                    Cb[(size_t)(rbase + r) * N + c + n * 16] = (bf16_t)acc[t][n][r];
        }
    }
}

// ---------------------------------------------------------------------------
// Repo tail: raw[row] = softplus( dot(gelu(H[row]+b1), W2) + b2 )
// ---------------------------------------------------------------------------
__global__ void repo_reduce_kernel(const float* __restrict__ H,
                                   const float* __restrict__ b1,
                                   const float* __restrict__ W2,
                                   const float* __restrict__ b2,
                                   float* __restrict__ raw) {
    __shared__ float red[POS_DIM];
    int row = blockIdx.x;
    int t = threadIdx.x;
    float x = H[(size_t)row * POS_DIM + t] + b1[t];
    float x3 = x * x * x;  // gelu, tanh approximation (jax default)
    float g = 0.5f * x * (1.0f + tanhf(0.7978845608028654f * (x + 0.044715f * x3)));
    red[t] = g * W2[t];
    __syncthreads();
    for (int s = 64; s > 0; s >>= 1) {
        if (t < s) red[t] += red[t + s];
        __syncthreads();
    }
    if (t == 0) {
        float r = red[0] + b2[0];
        raw[row] = (r > 20.0f) ? r : log1pf(expf(r));  // softplus
    }
}

// ---------------------------------------------------------------------------
// Inclusive cumsum of 2048 elements per batch (1024 threads, 2 elems/thread)
// ---------------------------------------------------------------------------
__global__ void cumsum_kernel(const float* __restrict__ raw, float* __restrict__ pos) {
    __shared__ float sc[1024];
    int b = blockIdx.x;
    int t = threadIdx.x;
    const float* r = raw + (size_t)b * SEQ;
    float a0 = r[2 * t], a1 = r[2 * t + 1];
    sc[t] = a0 + a1;
    __syncthreads();
    for (int off = 1; off < 1024; off <<= 1) {
        float v = (t >= off) ? sc[t - off] : 0.0f;
        __syncthreads();
        sc[t] += v;
        __syncthreads();
    }
    float excl = (t > 0) ? sc[t - 1] : 0.0f;
    pos[(size_t)b * SEQ + 2 * t]     = excl + a0;
    pos[(size_t)b * SEQ + 2 * t + 1] = excl + a0 + a1;
}

// ---------------------------------------------------------------------------
// RoPE in-place on bf16 Q and K. Q additionally scaled by 1/sqrt(64)=0.125
// (power of two -> exact in bf16), so attention needs no score scaling.
// ---------------------------------------------------------------------------
__global__ void rope_kernel(bf16_t* __restrict__ Q, bf16_t* __restrict__ Kb,
                            const float* __restrict__ pos) {
    int idx = blockIdx.x * blockDim.x + threadIdx.x;
    int d   = idx & 31;
    int h   = (idx >> 5) & (HEADS - 1);
    int tok = idx >> 9;
    if (tok >= TOKENS) return;
    float p = pos[tok];
    float ang = p * __expf(-0.28782313662425574f * (float)d);  // 10000^(-d/32)
    float c = __cosf(ang), s = __sinf(ang);
    size_t base = (size_t)tok * HIDDEN + h * HEAD_DIM;
    float q1 = (float)Q[base + d], q2 = (float)Q[base + d + 32];
    Q[base + d]      = (bf16_t)(0.125f * (q1 * c - q2 * s));
    Q[base + d + 32] = (bf16_t)(0.125f * (q1 * s + q2 * c));
    float k1 = (float)Kb[base + d], k2 = (float)Kb[base + d + 32];
    Kb[base + d]      = (bf16_t)(k1 * c - k2 * s);
    Kb[base + d + 32] = (bf16_t)(k1 * s + k2 * c);
}

// ---------------------------------------------------------------------------
// Flash attention: grid = (SEQ/128, BATCH*HEADS), block = 128 (4 waves).
// Each wave owns 32 query rows (2 row-tiles); key tiles of 32 with online
// softmax. K and V fragments are loaded once per key tile and reused by both
// row-tiles -> 16 b128 loads per 16 WMMAs (1 load/WMMA).
// Vt is per-batch transposed V: Vt[(b*HIDDEN + h*64 + d)*SEQ + s].
// ---------------------------------------------------------------------------
__global__ void __launch_bounds__(128)
attention_kernel(const bf16_t* __restrict__ Q,
                 const bf16_t* __restrict__ Kb,
                 const bf16_t* __restrict__ Vt,
                 bf16_t* __restrict__ O) {
    __shared__ __align__(16) bf16_t pshare[4][32 * 32];
    int lane = threadIdx.x & 31;
    int wave = threadIdx.x >> 5;
    int bh = blockIdx.y;
    int b = bh >> 4, h = bh & 15;
    int qrow0 = b * SEQ + blockIdx.x * 128 + wave * 32;
    int krow0 = b * SEQ;
    const int ld = HIDDEN;
    const bf16_t* Qp = Q + h * HEAD_DIM;
    const bf16_t* Kp = Kb + h * HEAD_DIM;
    const bf16_t* Vp = Vt + (size_t)(b * HIDDEN + h * HEAD_DIM) * SEQ;
    bf16_t* pbuf = pshare[wave];

    v16bf qa[2][2];  // [row tile][d half]
#pragma unroll
    for (int t = 0; t < 2; ++t) {
        qa[t][0] = load_a_frag(Qp, ld, qrow0 + t * 16, 0, lane);
        qa[t][1] = load_a_frag(Qp, ld, qrow0 + t * 16, 32, lane);
    }

    v8f acc[2][4] = {};
    float m[2][8], l[2][8];
#pragma unroll
    for (int t = 0; t < 2; ++t)
#pragma unroll
        for (int r = 0; r < 8; ++r) { m[t][r] = -1e30f; l[t][r] = 0.0f; }

    for (int j = 0; j < SEQ; j += 32) {
        // ---- K fragments: [key col group g][d half], shared by both row tiles
        v16bf kf[2][2];
#pragma unroll
        for (int g = 0; g < 2; ++g) {
            const bf16_t* kp = Kp + (size_t)(krow0 + j + g * 16 + (lane & 15)) * ld
                                  + ((lane >> 4) << 4);
            kf[g][0] = concat8(*(const v8bf*)kp,        *(const v8bf*)(kp + 8));
            kf[g][1] = concat8(*(const v8bf*)(kp + 32), *(const v8bf*)(kp + 40));
        }

        // ---- scores: 8 WMMAs (2 row tiles x 2 key groups x 2 d halves) ----
        v8f s[2][2];
#pragma unroll
        for (int t = 0; t < 2; ++t)
#pragma unroll
            for (int g = 0; g < 2; ++g) {
                v8f z = {};
                z = __builtin_amdgcn_wmma_f32_16x16x32_bf16(false, qa[t][0], false, kf[g][0],
                                                            (short)0, z, false, false);
                z = __builtin_amdgcn_wmma_f32_16x16x32_bf16(false, qa[t][1], false, kf[g][1],
                                                            (short)0, z, false, false);
                s[t][g] = z;
            }

        // ---- online softmax per row tile ----
#pragma unroll
        for (int t = 0; t < 2; ++t) {
            float alpha[8];
#pragma unroll
            for (int r = 0; r < 8; ++r) {
                float v0 = s[t][0][r], v1 = s[t][1][r];
                float mx = fmaxf(v0, v1);
#pragma unroll
                for (int off = 8; off >= 1; off >>= 1)
                    mx = fmaxf(mx, __shfl_xor(mx, off, 16));
                float mnew = fmaxf(m[t][r], mx);
                alpha[r] = __expf(m[t][r] - mnew);
                m[t][r] = mnew;
                float p0 = __expf(v0 - mnew);
                float p1 = __expf(v1 - mnew);
                s[t][0][r] = p0; s[t][1][r] = p1;
                float rs = p0 + p1;
#pragma unroll
                for (int off = 8; off >= 1; off >>= 1)
                    rs += __shfl_xor(rs, off, 16);
                l[t][r] = l[t][r] * alpha[r] + rs;
            }
#pragma unroll
            for (int n = 0; n < 4; ++n)
#pragma unroll
                for (int r = 0; r < 8; ++r)
                    acc[t][n][r] *= alpha[r];
        }

        // ---- stage P through per-wave LDS: C layout -> A layout ----
        {
            int c = lane & 15, rb = (lane >> 4) << 3;
#pragma unroll
            for (int t = 0; t < 2; ++t)
#pragma unroll
                for (int r = 0; r < 8; ++r) {
                    pbuf[(t * 16 + rb + r) * 32 + c]      = (bf16_t)s[t][0][r];
                    pbuf[(t * 16 + rb + r) * 32 + 16 + c] = (bf16_t)s[t][1][r];
                }
        }
        v16bf pa[2];  // same-wave DS ops are in order -> no barrier needed
#pragma unroll
        for (int t = 0; t < 2; ++t) {
            const bf16_t* pp = pbuf + (t * 16 + (lane & 15)) * 32 + ((lane >> 4) << 4);
            pa[t] = concat8(*(const v8bf*)pp, *(const v8bf*)(pp + 8));  // ds_load_b128 x2
        }

        // ---- V fragments (shared by both row tiles), then 8 PV WMMAs ----
        {
            const bf16_t* vbase = Vp + (size_t)(lane & 15) * SEQ + j + ((lane >> 4) << 4);
            v16bf vf[4];
#pragma unroll
            for (int n = 0; n < 4; ++n) {
                const bf16_t* vp = vbase + (size_t)(n * 16) * SEQ;
                vf[n] = concat8(*(const v8bf*)vp, *(const v8bf*)(vp + 8));
            }
#pragma unroll
            for (int t = 0; t < 2; ++t)
#pragma unroll
                for (int n = 0; n < 4; ++n)
                    acc[t][n] = __builtin_amdgcn_wmma_f32_16x16x32_bf16(false, pa[t], false, vf[n],
                                                                        (short)0, acc[t][n], false, false);
        }
    }

    // ---- normalize and store context (bf16 feeds the W_o GEMM) ----
    int c = lane & 15, rb = (lane >> 4) << 3;
#pragma unroll
    for (int t = 0; t < 2; ++t)
#pragma unroll
        for (int r = 0; r < 8; ++r) {
            float inv = 1.0f / l[t][r];
            bf16_t* op = O + (size_t)(qrow0 + t * 16 + rb + r) * HIDDEN + h * HEAD_DIM;
#pragma unroll
            for (int n = 0; n < 4; ++n)
                op[n * 16 + c] = (bf16_t)(acc[t][n][r] * inv);
        }
}

// ---------------------------------------------------------------------------
// Host-side orchestration
// ---------------------------------------------------------------------------
extern "C" void kernel_launch(void* const* d_in, const int* in_sizes, int n_in,
                              void* d_out, int out_size, void* d_ws, size_t ws_size,
                              hipStream_t stream) {
    const float* x       = (const float*)d_in[0];
    const float* W_q     = (const float*)d_in[1];
    const float* W_k     = (const float*)d_in[2];
    const float* W_v     = (const float*)d_in[3];
    const float* W_o     = (const float*)d_in[4];
    const float* repo_W1 = (const float*)d_in[5];
    const float* repo_b1 = (const float*)d_in[6];
    const float* repo_W2 = (const float*)d_in[7];
    const float* repo_b2 = (const float*)d_in[8];
    float* out = (float*)d_out;

    char* ws = (char*)d_ws;
    size_t off = 0;
    auto alloc = [&](size_t bytes) -> void* {
        void* p = ws + off;
        off += (bytes + 255) & ~(size_t)255;
        return p;
    };
    bf16_t* xb   = (bf16_t*)alloc((size_t)TOKENS * HIDDEN * 2);
    bf16_t* wqt  = (bf16_t*)alloc((size_t)HIDDEN * HIDDEN * 2);   // W^T (NxK)
    bf16_t* wkt  = (bf16_t*)alloc((size_t)HIDDEN * HIDDEN * 2);
    bf16_t* wvt  = (bf16_t*)alloc((size_t)HIDDEN * HIDDEN * 2);
    bf16_t* wot  = (bf16_t*)alloc((size_t)HIDDEN * HIDDEN * 2);
    bf16_t* w1t  = (bf16_t*)alloc((size_t)POS_DIM * HIDDEN * 2);  // W1^T (128x1024)
    bf16_t* qb   = (bf16_t*)alloc((size_t)TOKENS * HIDDEN * 2);
    bf16_t* kb   = (bf16_t*)alloc((size_t)TOKENS * HIDDEN * 2);
    bf16_t* vb   = (bf16_t*)alloc((size_t)TOKENS * HIDDEN * 2);
    bf16_t* vt   = (bf16_t*)alloc((size_t)TOKENS * HIDDEN * 2);   // per-batch V^T
    float*  hbuf = (float*)alloc((size_t)TOKENS * POS_DIM * 4);
    float*  raw  = (float*)alloc((size_t)TOKENS * 4);
    float*  pos  = (float*)alloc((size_t)TOKENS * 4);
    bf16_t* ctx  = (bf16_t*)alloc((size_t)TOKENS * HIDDEN * 2);
    (void)ws_size;

    // 1) x -> bf16 (vectorized); weights -> transposed bf16 (fused cvt)
    {
        int n8 = TOKENS * HIDDEN / 8;
        f32_to_bf16_vec_kernel<<<(n8 + 255) / 256, 256, 0, stream>>>(x, xb, n8);
    }
    {
        dim3 blk(256, 1, 1);
        dim3 gW(HIDDEN / 32, HIDDEN / 32, 1);   // in 1024x1024
        transpose_f32_to_bf16_kernel<<<gW, blk, 0, stream>>>(W_q, wqt, HIDDEN, HIDDEN);
        transpose_f32_to_bf16_kernel<<<gW, blk, 0, stream>>>(W_k, wkt, HIDDEN, HIDDEN);
        transpose_f32_to_bf16_kernel<<<gW, blk, 0, stream>>>(W_v, wvt, HIDDEN, HIDDEN);
        transpose_f32_to_bf16_kernel<<<gW, blk, 0, stream>>>(W_o, wot, HIDDEN, HIDDEN);
        dim3 gW1(POS_DIM / 32, HIDDEN / 32, 1); // in 1024x128
        transpose_f32_to_bf16_kernel<<<gW1, blk, 0, stream>>>(repo_W1, w1t, HIDDEN, POS_DIM);
    }

    // 2) Q/K/V projections (bf16 out)
    dim3 gblk(128, 1, 1);
    dim3 gproj(TOKENS / 128, HIDDEN / 64, 1);
    gemm_bf16_kernel<<<gproj, gblk, 0, stream>>>(xb, wqt, nullptr, qb, TOKENS, HIDDEN, HIDDEN);
    gemm_bf16_kernel<<<gproj, gblk, 0, stream>>>(xb, wkt, nullptr, kb, TOKENS, HIDDEN, HIDDEN);
    gemm_bf16_kernel<<<gproj, gblk, 0, stream>>>(xb, wvt, nullptr, vb, TOKENS, HIDDEN, HIDDEN);

    // 3) repo positions: H = x @ W1 (f32), reduce, per-batch cumsum
    dim3 gpos(TOKENS / 128, POS_DIM / 64, 1);
    gemm_bf16_kernel<<<gpos, gblk, 0, stream>>>(xb, w1t, hbuf, nullptr, TOKENS, POS_DIM, HIDDEN);
    repo_reduce_kernel<<<TOKENS, POS_DIM, 0, stream>>>(hbuf, repo_b1, repo_W2, repo_b2, raw);
    cumsum_kernel<<<BATCH, 1024, 0, stream>>>(raw, pos);

    // 4) RoPE on Q (scaled by 0.125) and K
    {
        int n = TOKENS * HEADS * 32;
        rope_kernel<<<(n + 255) / 256, 256, 0, stream>>>(qb, kb, pos);
    }

    // 5) per-batch V transpose: (SEQ x HIDDEN) -> (HIDDEN x SEQ)
    {
        dim3 blk(256, 1, 1);
        dim3 gV(HIDDEN / 32, SEQ / 32, 1);
        for (int b = 0; b < BATCH; ++b)
            transpose_bf16_kernel<<<gV, blk, 0, stream>>>(
                vb + (size_t)b * SEQ * HIDDEN, vt + (size_t)b * HIDDEN * SEQ, SEQ, HIDDEN);
    }

    // 6) attention -> ctx (bf16)
    dim3 gatt(SEQ / 128, BATCH * HEADS, 1);
    attention_kernel<<<gatt, gblk, 0, stream>>>(qb, kb, vt, ctx);

    // 7) output projection -> d_out (f32)
    gemm_bf16_kernel<<<gproj, gblk, 0, stream>>>(ctx, wot, out, nullptr, TOKENS, HIDDEN, HIDDEN);
    (void)out_size; (void)n_in; (void)in_sizes;
}
